// ANI_89249420411567
// MI455X (gfx1250) — matrix operand order repaired and checked by reading, not compiled, not run on hardware
//
#include <hip/hip_runtime.h>

// ---------------------------------------------------------------------------
// ANI-style 4-expert MLP (384 -> 192 -> 160 -> 1, CELU) over 262144 atoms,
// per-atom expert dispatch by type, global scalar energy sum.
//
// Strategy for MI455X (gfx1250, wave32, WMMA):
//   * gather atoms by type (avoid 4x expert waste -> stay HBM-bound)
//   * bf16 WMMA 16x16x32 with f32 accumulation for both GEMM layers
//   * weights converted to bf16 once, transposed so B-fragments are
//     contiguous 16B-per-lane loads
//   * X tile staged f32->bf16 into LDS; A-fragments read from LDS per the
//     ISA 16-bit A-matrix layout (two 8-elem runs per lane)
// ---------------------------------------------------------------------------

typedef __attribute__((ext_vector_type(16))) __bf16 v16bf;
typedef __attribute__((ext_vector_type(8)))  __bf16 v8bf;
typedef __attribute__((ext_vector_type(4)))  __bf16 v4bf;
typedef __attribute__((ext_vector_type(8)))  float  v8f;

#define IN_DIM  384
#define H1_DIM  192
#define H2_DIM  160
#define N_TYPES 4
#define N_ATOMS 262144
#define TILE_M  32          // atoms per block
#define NWAVES  4           // 128 threads

// ----- WMMA fragment loaders -----------------------------------------------
// A matrix 16x32 bf16 (ISA 7.12.2): lane L<16 holds row M=L, runs K={k0..k0+7}
// and K={k0+16..k0+23}; lane L>=16 holds row M=L-16 with runs shifted by +8.
__device__ __forceinline__ v16bf load_a_frag(const __bf16* base, int ldk,
                                             int m0, int k0, int lane) {
  const int hs  = lane >> 4;            // half-select
  const int row = m0 + (lane & 15);
  const __bf16* p = base + row * ldk + k0 + hs * 8;
  v8bf lo = *(const v8bf*)(p);          // 16B LDS load
  v8bf hi = *(const v8bf*)(p + 16);     // 16B LDS load
  v16bf r;
#pragma unroll
  for (int i = 0; i < 8; ++i) { r[i] = lo[i]; r[i + 8] = hi[i]; }
  return r;
}

// B matrix 32x16 bf16: lane L<16 holds column N=n0+L, K={k0..k0+7} and
// K={k0+16..k0+23}; lane L>=16 holds column N=n0+L-16 with runs shifted by +8.
// wt is the transposed weight matrix, layout [N][K] row-major -> contiguous.
__device__ __forceinline__ v16bf load_b_frag(const __bf16* wt, int K,
                                             int n0, int k0, int lane) {
  const int hs = lane >> 4;
  const int n  = n0 + (lane & 15);
  const __bf16* p = wt + n * K + k0 + hs * 8;
  v8bf lo = *(const v8bf*)(p);          // global_load_b128
  v8bf hi = *(const v8bf*)(p + 16);     // global_load_b128
  v16bf r;
#pragma unroll
  for (int i = 0; i < 8; ++i) { r[i] = lo[i]; r[i + 8] = hi[i]; }
  return r;
}

__device__ __forceinline__ float celu1(float x) {
  return x > 0.0f ? x : expm1f(x);
}

// ----- kernel 1: convert + transpose weights to bf16 ------------------------
__global__ __launch_bounds__(256)
void k_cvt_weights(const float* __restrict__ W1, const float* __restrict__ W2,
                   __bf16* __restrict__ Wt1, __bf16* __restrict__ Wt2) {
  int i = blockIdx.x * 256 + threadIdx.x;
  if (i < N_TYPES * IN_DIM * H1_DIM) {           // W1[e][k][n] -> Wt1[e][n][k]
    int e = i / (IN_DIM * H1_DIM), r = i % (IN_DIM * H1_DIM);
    int k = r / H1_DIM, n = r % H1_DIM;
    Wt1[(e * H1_DIM + n) * IN_DIM + k] = (__bf16)W1[i];
  }
  if (i < N_TYPES * H1_DIM * H2_DIM) {           // W2[e][k][n] -> Wt2[e][n][k]
    int e = i / (H1_DIM * H2_DIM), r = i % (H1_DIM * H2_DIM);
    int k = r / H2_DIM, n = r % H2_DIM;
    Wt2[(e * H2_DIM + n) * H1_DIM + k] = (__bf16)W2[i];
  }
}

// ----- kernel 2: bucket atom indices by expert type -------------------------
__global__ __launch_bounds__(256)
void k_build_index(const int* __restrict__ types, int* __restrict__ counts,
                   int* __restrict__ idx) {
  int i = blockIdx.x * 256 + threadIdx.x;
  if (i < N_ATOMS) {
    int e = types[i] & 3;
    int slot = atomicAdd(&counts[e], 1);
    idx[e * N_ATOMS + slot] = i;
  }
}

// ----- kernel 3: WMMA MLP over one 32-atom tile of one expert ---------------
__global__ __launch_bounds__(NWAVES * 32)
void k_mlp(const float* __restrict__ aev,
           const int* __restrict__ counts, const int* __restrict__ idx,
           const __bf16* __restrict__ Wt1, const float* __restrict__ b1,
           const __bf16* __restrict__ Wt2, const float* __restrict__ b2,
           const float* __restrict__ W3, const float* __restrict__ b3,
           float* __restrict__ out) {
  __shared__ __bf16 Xs[TILE_M * IN_DIM];    // 24 KB, bf16 row-major
  __shared__ __bf16 H1s[TILE_M * H1_DIM];   // 12 KB
  __shared__ int sIdx[TILE_M];

  const int e    = blockIdx.y;
  const int cnt  = counts[e];
  const int base = blockIdx.x * TILE_M;
  if (base >= cnt) return;                  // uniform early exit
  const int mCount = min(TILE_M, cnt - base);
  const int tid  = threadIdx.x;
  const int wave = tid >> 5;
  const int lane = tid & 31;

  if (tid < TILE_M)
    sIdx[tid] = (tid < mCount) ? idx[e * N_ATOMS + base + tid] : -1;
  __syncthreads();

  // ---- stage X tile: gather rows, f32 -> bf16, zero-fill pad rows ----
  // 32 rows x 96 float4 chunks = 3072 chunks, 24 per thread
  for (int c = tid; c < TILE_M * (IN_DIM / 4); c += NWAVES * 32) {
    int row = c / (IN_DIM / 4);
    int q   = c % (IN_DIM / 4);
    float4 v = make_float4(0.f, 0.f, 0.f, 0.f);
    if (row < mCount) {
      const float4* src = (const float4*)(aev + (size_t)sIdx[row] * IN_DIM);
      v = src[q];
    }
    v4bf b; b[0] = (__bf16)v.x; b[1] = (__bf16)v.y;
            b[2] = (__bf16)v.z; b[3] = (__bf16)v.w;
    *(v4bf*)&Xs[row * IN_DIM + q * 4] = b;
  }
  __syncthreads();

  const __bf16* wt1 = Wt1 + (size_t)e * H1_DIM * IN_DIM;
  const __bf16* wt2 = Wt2 + (size_t)e * H2_DIM * H1_DIM;
  const int n_in_lane = lane & 15;
  const int hs        = lane >> 4;

  // ---- layer 1: [32 x 384] x [384 x 192], N-tiles split across waves ----
  for (int nt = wave; nt < H1_DIM / 16; nt += NWAVES) {
    const int n0 = nt * 16;
    v8f acc0 = {}; v8f acc1 = {};
    for (int kt = 0; kt < IN_DIM / 32; ++kt) {
      const int k0 = kt * 32;
      v16bf bf = load_b_frag(wt1, IN_DIM, n0, k0, lane);
      v16bf a0 = load_a_frag(Xs, IN_DIM,  0, k0, lane);
      v16bf a1 = load_a_frag(Xs, IN_DIM, 16, k0, lane);
      acc0 = __builtin_amdgcn_wmma_f32_16x16x32_bf16(false, a0, false, bf,
                                                     (short)0, acc0, false, false);
      acc1 = __builtin_amdgcn_wmma_f32_16x16x32_bf16(false, a1, false, bf,
                                                     (short)0, acc1, false, false);
    }
    const int  n    = n0 + n_in_lane;
    const float bias = b1[e * H1_DIM + n];
#pragma unroll
    for (int r = 0; r < 8; ++r) {
      float v0 = celu1(acc0[r] + bias);
      float v1 = celu1(acc1[r] + bias);
      H1s[(r + 8 * hs) * H1_DIM + n]        = (__bf16)v0;   // M-frag 0
      H1s[(16 + r + 8 * hs) * H1_DIM + n]   = (__bf16)v1;   // M-frag 1
    }
  }
  __syncthreads();

  // ---- layer 2: [32 x 192] x [192 x 160], fused layer-3 dot vs w3 ----
  float lacc = 0.0f;
  for (int nt = wave; nt < H2_DIM / 16; nt += NWAVES) {
    const int n0 = nt * 16;
    v8f acc0 = {}; v8f acc1 = {};
    for (int kt = 0; kt < H1_DIM / 32; ++kt) {
      const int k0 = kt * 32;
      v16bf bf = load_b_frag(wt2, H1_DIM, n0, k0, lane);
      v16bf a0 = load_a_frag(H1s, H1_DIM,  0, k0, lane);
      v16bf a1 = load_a_frag(H1s, H1_DIM, 16, k0, lane);
      acc0 = __builtin_amdgcn_wmma_f32_16x16x32_bf16(false, a0, false, bf,
                                                     (short)0, acc0, false, false);
      acc1 = __builtin_amdgcn_wmma_f32_16x16x32_bf16(false, a1, false, bf,
                                                     (short)0, acc1, false, false);
    }
    const int   n    = n0 + n_in_lane;
    const float bias = b2[e * H2_DIM + n];
    const float w3v  = W3[e * H2_DIM + n];     // W3 is [E][160][1]
#pragma unroll
    for (int r = 0; r < 8; ++r) {
      float v0 = celu1(acc0[r] + bias);
      float v1 = celu1(acc1[r] + bias);
      int m0i = r + 8 * hs;
      int m1i = 16 + r + 8 * hs;
      lacc += (m0i < mCount) ? v0 * w3v : 0.0f;   // mask pad rows
      lacc += (m1i < mCount) ? v1 * w3v : 0.0f;
    }
  }

  // ---- wave reduction + global scalar accumulation ----
#pragma unroll
  for (int off = 16; off > 0; off >>= 1)
    lacc += __shfl_down(lacc, off, 32);
  if (lane == 0) atomicAdd(out, lacc);
  if (tid == 0)  atomicAdd(out, b3[e] * (float)mCount);   // per-atom bias
}

// ---------------------------------------------------------------------------
extern "C" void kernel_launch(void* const* d_in, const int* in_sizes, int n_in,
                              void* d_out, int out_size, void* d_ws, size_t ws_size,
                              hipStream_t stream) {
  const float* aev   = (const float*)d_in[0];
  const int*   types = (const int*)d_in[1];
  const float* W1    = (const float*)d_in[2];
  const float* b1    = (const float*)d_in[3];
  const float* W2    = (const float*)d_in[4];
  const float* b2    = (const float*)d_in[5];
  const float* W3    = (const float*)d_in[6];
  const float* b3    = (const float*)d_in[7];
  float* out = (float*)d_out;

  // scratch layout
  char* ws = (char*)d_ws;
  int*    counts = (int*)ws;                                        // 16 B
  int*    idx    = (int*)(ws + 256);                                // 4 MB
  __bf16* Wt1    = (__bf16*)(ws + 256 + (size_t)N_TYPES * N_ATOMS * 4);
  __bf16* Wt2    = Wt1 + (size_t)N_TYPES * H1_DIM * IN_DIM;

  hipMemsetAsync(out, 0, sizeof(float), stream);
  hipMemsetAsync(counts, 0, 256, stream);

  k_cvt_weights<<<(N_TYPES * IN_DIM * H1_DIM + 255) / 256, 256, 0, stream>>>(
      W1, W2, Wt1, Wt2);
  k_build_index<<<(N_ATOMS + 255) / 256, 256, 0, stream>>>(types, counts, idx);

  dim3 grid((N_ATOMS + TILE_M - 1) / TILE_M, N_TYPES);   // over-provisioned;
  k_mlp<<<grid, NWAVES * 32, 0, stream>>>(               // blocks early-exit
      aev, counts, idx, Wt1, b1, Wt2, b2, W3, b3, out);
}
